// SimpleGeoFormerModel_37220186587482
// MI455X (gfx1250) — matrix operand order repaired and checked by reading, not compiled, not run on hardware
//
#include <hip/hip_runtime.h>
#include <math.h>

#define B_  8
#define N_  256
#define E_  512
#define H_  8
#define DH_ 64
#define FF_ 2048
#define L_  4
#define E3_ 1536

typedef unsigned short u16;
typedef __attribute__((ext_vector_type(16))) __bf16 bf16x16;
typedef __attribute__((ext_vector_type(8)))  float  f32x8;
typedef __attribute__((ext_vector_type(4)))  int    i32x4;

// Address-space-qualified vector types for the async global->LDS builtin.
// (Attributes attached via typedef so they bind to the pointee type.)
typedef __attribute__((address_space(1))) i32x4 as1_i32x4;
typedef __attribute__((address_space(3))) i32x4 as3_i32x4;

// ---- CDNA5 async global->LDS path (guarded; falls back to VGPR round-trip) ----
#if defined(__gfx1250__) && __has_builtin(__builtin_amdgcn_global_load_async_to_lds_b128)
#define HAVE_ASYNC_LDS 1
#else
#define HAVE_ASYNC_LDS 0
#endif

#if HAVE_ASYNC_LDS
#if __has_builtin(__builtin_amdgcn_s_wait_asynccnt)
#define WAIT_ASYNC() __builtin_amdgcn_s_wait_asynccnt(0)
#else
#define WAIT_ASYNC() asm volatile("s_wait_asynccnt 0x0" ::: "memory")
#endif
#else
#define WAIT_ASYNC() ((void)0)
#endif

__device__ __forceinline__ void async_b128(const u16* g, u16* l) {
#if HAVE_ASYNC_LDS
    __builtin_amdgcn_global_load_async_to_lds_b128(
        (as1_i32x4*)(u16*)g, (as3_i32x4*)l, 0, 0);
#else
    *(uint4*)l = *(const uint4*)g;
#endif
}

union FragU { uint4 u[2]; bf16x16 v; };

__device__ __forceinline__ u16 f2bf(float f) {
    unsigned int u = __float_as_uint(f);
    u += 0x7FFFu + ((u >> 16) & 1u);
    return (u16)(u >> 16);
}

__device__ __forceinline__ f32x8 wmma_bf16(uint4 a0, uint4 a1, uint4 b0, uint4 b1, f32x8 c) {
    FragU a, b;
    a.u[0] = a0; a.u[1] = a1;
    b.u[0] = b0; b.u[1] = b1;
    return __builtin_amdgcn_wmma_f32_16x16x32_bf16(false, a.v, false, b.v,
                                                   (short)0, c, false, false);
}

// ---------------------------------------------------------------------------
// fp32 -> bf16 conversion
// ---------------------------------------------------------------------------
__global__ void k_f2bf(const float* __restrict__ in, u16* __restrict__ out, int n) {
    int i = blockIdx.x * 256 + threadIdx.x;
    if (i < n) out[i] = f2bf(in[i]);
}

// ---------------------------------------------------------------------------
// Embedding gather: x (f32 master) + x (bf16 gemm input)
// ---------------------------------------------------------------------------
__global__ __launch_bounds__(256) void k_embed(const int* __restrict__ tok,
                                               const float* __restrict__ emb,
                                               float* __restrict__ xf,
                                               u16* __restrict__ xb) {
    int row = blockIdx.x;
    int t = tok[row];
#pragma unroll
    for (int it = 0; it < 2; ++it) {
        int e = threadIdx.x + it * 256;
        float v = emb[(size_t)t * E_ + e];
        xf[(size_t)row * E_ + e] = v;
        xb[(size_t)row * E_ + e] = f2bf(v);
    }
}

// ---------------------------------------------------------------------------
// Distance-projection attention bias: bias[b,q,k] = sum_e tanh(d*w1+b1)*w2 + b2
// ---------------------------------------------------------------------------
__global__ __launch_bounds__(256) void k_bias(const float* __restrict__ dist,
                                              const int* __restrict__ tok,
                                              const float* __restrict__ w1,
                                              const float* __restrict__ bb1,
                                              const float* __restrict__ w2,
                                              const float* __restrict__ b2s,
                                              float* __restrict__ biasf) {
    __shared__ float sw1[E_], sb1[E_], sw2[E_];
    int q = blockIdx.x, b = blockIdx.y;
#pragma unroll
    for (int it = 0; it < 2; ++it) {
        int e = threadIdx.x + it * 256;
        sw1[e] = w1[e]; sb1[e] = bb1[e]; sw2[e] = w2[e];
    }
    __syncthreads();
    int k = threadIdx.x;
    float d = dist[((size_t)(b * N_ + q)) * N_ + k];
    float acc = 0.f;
    for (int e = 0; e < E_; ++e)
        acc += tanhf(fmaf(d, sw1[e], sb1[e])) * sw2[e];
    acc += b2s[0];
    bool pad = (tok[b * N_ + k] == 0) || (tok[b * N_ + q] == 0);
    biasf[((size_t)(b * N_ + q)) * N_ + k] = pad ? -INFINITY : acc;
}

// ---------------------------------------------------------------------------
// Generic bf16 WMMA GEMM: C[M,Nc] = A[M,K] @ B[Nc,K]^T (+bias) (+gelu)
// BM=BN=128, BK=32; 8 waves x 32x64 tiles. Double-buffered async LDS staging.
// mode: 0 = f32 out, 1 = bf16 out, 2 = gelu(exact erf) -> bf16 out
// ---------------------------------------------------------------------------
__global__ __launch_bounds__(256) void k_gemm(const u16* __restrict__ A,
                                              const u16* __restrict__ Bw,
                                              const float* __restrict__ bias,
                                              float* __restrict__ outf,
                                              u16* __restrict__ outb,
                                              int M, int Nc, int K, int mode) {
    __shared__ __align__(16) u16 Ash[2][128 * 40];
    __shared__ __align__(16) u16 Bsh[2][128 * 40];
    const int tid  = threadIdx.x;
    const int lane = tid & 31, wid = tid >> 5;
    const int wm = wid & 3, wn = wid >> 2;
    const int m0 = blockIdx.y * 128, n0 = blockIdx.x * 128;
    const int row = tid >> 2, cg = tid & 3;
    const int ln  = lane & 15;
    const int kb  = (lane < 16) ? 0 : 16;  // A-frag K base (contiguous k = 16h + s)
    const int kb2 = (lane < 16) ? 0 : 16;  // B-frag K base (contiguous k = 16h + s, same as A)
    f32x8 acc[2][4] = {};
    (void)M;

    // stage one 128x32 A tile + 128x32 B tile into buffer `buf`
    auto stage = [&](int buf, int kt) {
#pragma unroll
        for (int p = 0; p < 2; ++p) {
            int r = row + p * 64;
            async_b128(&A [(size_t)(m0 + r) * K + kt + cg * 8], &Ash[buf][r * 40 + cg * 8]);
            async_b128(&Bw[(size_t)(n0 + r) * K + kt + cg * 8], &Bsh[buf][r * 40 + cg * 8]);
        }
    };

    stage(0, 0);
    int buf = 0;
    for (int kt = 0; kt < K; kt += 32) {
        WAIT_ASYNC();            // this wave's staged tile is in LDS
        __syncthreads();         // all waves staged; all waves done reading buf^1
        if (kt + 32 < K) stage(buf ^ 1, kt + 32);   // overlap next tile with compute
        if (kt + 64 < K) {
            __builtin_prefetch(&A [(size_t)(m0 + row) * K + kt + 64], 0, 1);
            __builtin_prefetch(&Bw[(size_t)(n0 + row) * K + kt + 64], 0, 1);
        }
        const u16* Ab = Ash[buf];
        const u16* Bb = Bsh[buf];

        uint4 a0[2], a1[2];
#pragma unroll
        for (int s = 0; s < 2; ++s) {
            int m = wm * 32 + s * 16 + ln;
            a0[s] = *(const uint4*)&Ab[m * 40 + kb];
            a1[s] = *(const uint4*)&Ab[m * 40 + kb + 8];
        }
#pragma unroll
        for (int nt = 0; nt < 4; ++nt) {
            int n = wn * 64 + nt * 16 + ln;
            uint4 b0 = *(const uint4*)&Bb[n * 40 + kb2];
            uint4 b1 = *(const uint4*)&Bb[n * 40 + kb2 + 8];
#pragma unroll
            for (int s = 0; s < 2; ++s)
                acc[s][nt] = wmma_bf16(a0[s], a1[s], b0, b1, acc[s][nt]);
        }
        buf ^= 1;
    }

    const int halfofs = (lane >> 4) << 3;   // C layout: lanes 16-31 hold M+8
#pragma unroll
    for (int s = 0; s < 2; ++s)
#pragma unroll
        for (int nt = 0; nt < 4; ++nt) {
            int col = n0 + wn * 64 + nt * 16 + ln;
            float bv = bias ? bias[col] : 0.0f;
#pragma unroll
            for (int r = 0; r < 8; ++r) {
                int rw = m0 + wm * 32 + s * 16 + r + halfofs;
                float v = acc[s][nt][r] + bv;
                if (mode == 2) v = 0.5f * v * (1.0f + erff(v * 0.70710678118f));
                size_t idx = (size_t)rw * Nc + col;
                if (mode == 0) outf[idx] = v;
                else           outb[idx] = f2bf(v);
            }
        }
}

// ---------------------------------------------------------------------------
// Flash-style attention. One block per (b,h); 8 waves x 2 q-tiles of 16 rows.
// KV in two 128-key blocks: K staged async to LDS, V transposed via registers.
// ---------------------------------------------------------------------------
__global__ __launch_bounds__(256) void k_attn(const u16* __restrict__ qkv,
                                              const float* __restrict__ bias,
                                              u16* __restrict__ out) {
    __shared__ __align__(16) u16 Ksh[128 * 64];
    __shared__ __align__(16) u16 Vtsh[64 * 128];
    __shared__ __align__(16) u16 Psh[8][512];   // per-wave 16x32 prob scratch
    const int b = blockIdx.y, h = blockIdx.x;
    const int tid = threadIdx.x, lane = tid & 31, wid = tid >> 5;
    const int ln = lane & 15;
    const int kb  = (lane < 16) ? 0 : 16;
    const int kb2 = (lane < 16) ? 0 : 16;
    const int halfofs = (lane >> 4) << 3;
    const float scale = 0.125f;   // 1/sqrt(64)

    f32x8 o_acc[2][4] = {};
    float m_st[2][8], l_st[2][8];
#pragma unroll
    for (int qt = 0; qt < 2; ++qt)
#pragma unroll
        for (int r = 0; r < 8; ++r) { m_st[qt][r] = -INFINITY; l_st[qt][r] = 0.f; }

    for (int blk = 0; blk < 2; ++blk) {
        const int k_base = blk * 128;
        __syncthreads();
        // K block -> Ksh[128][64] (async); V block -> Vtsh[64][128] transposed
#pragma unroll
        for (int p = 0; p < 4; ++p) {
            int idx = tid + p * 256;
            int krow = idx >> 3, cg = idx & 7;
            size_t base = (size_t)(b * N_ + k_base + krow) * E3_;
            async_b128(&qkv[base + E_ + h * DH_ + cg * 8], &Ksh[krow * 64 + cg * 8]);
            uint4 vq = *(const uint4*)&qkv[base + 2 * E_ + h * DH_ + cg * 8];
            const u16* ve = (const u16*)&vq;
#pragma unroll
            for (int j = 0; j < 8; ++j)
                Vtsh[(cg * 8 + j) * 128 + krow] = ve[j];
        }
        WAIT_ASYNC();
        __syncthreads();

#pragma unroll
        for (int qt = 0; qt < 2; ++qt) {
            const int q0 = (wid * 2 + qt) * 16;
            f32x8 sc[8] = {};
            // scores = Q @ K^T over dh=64 (two K=32 wmma steps)
#pragma unroll
            for (int ks = 0; ks < 2; ++ks) {
                size_t qbase = (size_t)(b * N_ + q0 + ln) * E3_ + h * DH_ + ks * 32 + kb;
                uint4 a0 = *(const uint4*)&qkv[qbase];
                uint4 a1 = *(const uint4*)&qkv[qbase + 8];
#pragma unroll
                for (int nt = 0; nt < 8; ++nt) {
                    int n = nt * 16 + ln;
                    uint4 b0 = *(const uint4*)&Ksh[n * 64 + ks * 32 + kb2];
                    uint4 b1 = *(const uint4*)&Ksh[n * 64 + ks * 32 + kb2 + 8];
                    sc[nt] = wmma_bf16(a0, a1, b0, b1, sc[nt]);
                }
            }
            // scale + distance bias
#pragma unroll
            for (int nt = 0; nt < 8; ++nt)
#pragma unroll
                for (int r = 0; r < 8; ++r) {
                    int qr = q0 + r + halfofs;
                    int kc = k_base + nt * 16 + ln;
                    sc[nt][r] = sc[nt][r] * scale +
                                bias[((size_t)(b * N_ + qr)) * N_ + kc];
                }
            // online softmax update
            float nm[8], fac[8];
#pragma unroll
            for (int r = 0; r < 8; ++r) {
                float mx = -INFINITY;
#pragma unroll
                for (int nt = 0; nt < 8; ++nt) mx = fmaxf(mx, sc[nt][r]);
#pragma unroll
                for (int xm = 1; xm <= 8; xm <<= 1) mx = fmaxf(mx, __shfl_xor(mx, xm));
                float m2 = fmaxf(m_st[qt][r], mx);
                fac[r] = __expf(m_st[qt][r] - m2);
                m_st[qt][r] = m2;
                nm[r] = m2;
            }
#pragma unroll
            for (int nt = 0; nt < 8; ++nt)
#pragma unroll
                for (int r = 0; r < 8; ++r)
                    sc[nt][r] = __expf(sc[nt][r] - nm[r]);
#pragma unroll
            for (int r = 0; r < 8; ++r) {
                float s = 0.f;
#pragma unroll
                for (int nt = 0; nt < 8; ++nt) s += sc[nt][r];
#pragma unroll
                for (int xm = 1; xm <= 8; xm <<= 1) s += __shfl_xor(s, xm);
                l_st[qt][r] = l_st[qt][r] * fac[r] + s;
            }
#pragma unroll
            for (int dt = 0; dt < 4; ++dt)
#pragma unroll
                for (int r = 0; r < 8; ++r)
                    o_acc[qt][dt][r] *= fac[r];
            // P @ V: relayout probs C-format -> A-format via per-wave LDS scratch
#pragma unroll
            for (int pt = 0; pt < 4; ++pt) {
#pragma unroll
                for (int tt = 0; tt < 2; ++tt) {
                    int t = pt * 2 + tt;
#pragma unroll
                    for (int r = 0; r < 8; ++r)
                        Psh[wid][(r + halfofs) * 32 + tt * 16 + ln] = f2bf(sc[t][r]);
                }
                uint4 a0 = *(const uint4*)&Psh[wid][ln * 32 + kb];
                uint4 a1 = *(const uint4*)&Psh[wid][ln * 32 + kb + 8];
#pragma unroll
                for (int dt = 0; dt < 4; ++dt) {
                    uint4 b0 = *(const uint4*)&Vtsh[(dt * 16 + ln) * 128 + pt * 32 + kb2];
                    uint4 b1 = *(const uint4*)&Vtsh[(dt * 16 + ln) * 128 + pt * 32 + kb2 + 8];
                    o_acc[qt][dt] = wmma_bf16(a0, a1, b0, b1, o_acc[qt][dt]);
                }
            }
        }
    }
    // normalize and store bf16 attention output
#pragma unroll
    for (int qt = 0; qt < 2; ++qt) {
        const int q0 = (wid * 2 + qt) * 16;
#pragma unroll
        for (int dt = 0; dt < 4; ++dt)
#pragma unroll
            for (int r = 0; r < 8; ++r) {
                int qr = q0 + r + halfofs;
                float v = o_acc[qt][dt][r] / l_st[qt][r];
                out[(size_t)(b * N_ + qr) * E_ + h * DH_ + dt * 16 + ln] = f2bf(v);
            }
    }
}

// ---------------------------------------------------------------------------
// Fused residual + LayerNorm: x = LN(x + y); writes f32 master + bf16 copy.
// ---------------------------------------------------------------------------
__global__ __launch_bounds__(256) void k_ln(float* __restrict__ x,
                                            const float* __restrict__ y,
                                            const float* __restrict__ g,
                                            const float* __restrict__ bb,
                                            u16* __restrict__ xb) {
    __shared__ float red[18];
    int row = blockIdx.x, tid = threadIdx.x;
    int lane = tid & 31, wid = tid >> 5;
    float v[2], s = 0.f, sq = 0.f;
#pragma unroll
    for (int it = 0; it < 2; ++it) {
        int e = tid + it * 256;
        float t = x[(size_t)row * E_ + e] + y[(size_t)row * E_ + e];
        v[it] = t; s += t; sq += t * t;
    }
#pragma unroll
    for (int xm = 1; xm <= 16; xm <<= 1) { s += __shfl_xor(s, xm); sq += __shfl_xor(sq, xm); }
    if (lane == 0) { red[wid] = s; red[8 + wid] = sq; }
    __syncthreads();
    if (tid == 0) {
        float ts = 0.f, tq = 0.f;
        for (int i = 0; i < 8; ++i) { ts += red[i]; tq += red[8 + i]; }
        float mean = ts / (float)E_;
        float var  = tq / (float)E_ - mean * mean;
        red[16] = mean;
        red[17] = rsqrtf(var + 1e-5f);
    }
    __syncthreads();
    float mean = red[16], inv = red[17];
#pragma unroll
    for (int it = 0; it < 2; ++it) {
        int e = tid + it * 256;
        float o = (v[it] - mean) * inv * g[e] + bb[e];
        x[(size_t)row * E_ + e]  = o;
        xb[(size_t)row * E_ + e] = f2bf(o);
    }
}

// ---------------------------------------------------------------------------
// Final output: x (f32) then pad_mask (as 0/1 float)
// ---------------------------------------------------------------------------
__global__ void k_out(const float* __restrict__ x, const int* __restrict__ tok,
                      float* __restrict__ out) {
    int i = blockIdx.x * 256 + threadIdx.x;
    if (i < B_ * N_ * E_) out[i] = x[i];
    if (i < B_ * N_)      out[B_ * N_ * E_ + i] = (tok[i] == 0) ? 1.0f : 0.0f;
}

// ---------------------------------------------------------------------------
extern "C" void kernel_launch(void* const* d_in, const int* in_sizes, int n_in,
                              void* d_out, int out_size, void* d_ws, size_t ws_size,
                              hipStream_t stream) {
    (void)in_sizes; (void)n_in; (void)out_size; (void)ws_size;

    const int*   tok   = (const int*)d_in[0];
    const float* dist  = (const float*)d_in[2];
    const float* emb   = (const float*)d_in[4];
    const float* dp_w1 = (const float*)d_in[5];
    const float* dp_b1 = (const float*)d_in[6];
    const float* dp_w2 = (const float*)d_in[7];
    const float* dp_b2 = (const float*)d_in[8];
    const float* Wqkv  = (const float*)d_in[9];
    const float* bqkv  = (const float*)d_in[10];
    const float* Wo    = (const float*)d_in[11];
    const float* bo    = (const float*)d_in[12];
    const float* ln1g  = (const float*)d_in[13];
    const float* ln1b  = (const float*)d_in[14];
    const float* W1    = (const float*)d_in[15];
    const float* b1    = (const float*)d_in[16];
    const float* W2    = (const float*)d_in[17];
    const float* b2    = (const float*)d_in[18];
    const float* ln2g  = (const float*)d_in[19];
    const float* ln2b  = (const float*)d_in[20];

    char* ws = (char*)d_ws;
    size_t off = 0;
    auto alloc = [&](size_t bytes) -> char* {
        char* p = ws + off;
        off = (off + bytes + 255) & ~(size_t)255;
        return p;
    };
    u16*   wqkv_bf = (u16*)alloc((size_t)L_ * E3_ * E_ * 2);
    u16*   wo_bf   = (u16*)alloc((size_t)L_ * E_ * E_ * 2);
    u16*   w1_bf   = (u16*)alloc((size_t)L_ * FF_ * E_ * 2);
    u16*   w2_bf   = (u16*)alloc((size_t)L_ * E_ * FF_ * 2);
    float* xf      = (float*)alloc((size_t)B_ * N_ * E_ * 4);
    u16*   xbf     = (u16*)alloc((size_t)B_ * N_ * E_ * 2);
    u16*   qkvbf   = (u16*)alloc((size_t)B_ * N_ * E3_ * 2);
    u16*   attnbf  = (u16*)alloc((size_t)B_ * N_ * E_ * 2);
    u16*   hbf     = (u16*)alloc((size_t)B_ * N_ * FF_ * 2);
    float* yf      = (float*)alloc((size_t)B_ * N_ * E_ * 4);
    float* biasf   = (float*)alloc((size_t)B_ * N_ * N_ * 4);

    const int M = B_ * N_;   // 2048

    // weights -> bf16 (L2-resident for the whole pass)
    int n;
    n = L_ * E3_ * E_; k_f2bf<<<(n + 255) / 256, 256, 0, stream>>>(Wqkv, wqkv_bf, n);
    n = L_ * E_ * E_;  k_f2bf<<<(n + 255) / 256, 256, 0, stream>>>(Wo,   wo_bf,   n);
    n = L_ * FF_ * E_; k_f2bf<<<(n + 255) / 256, 256, 0, stream>>>(W1,   w1_bf,   n);
    n = L_ * E_ * FF_; k_f2bf<<<(n + 255) / 256, 256, 0, stream>>>(W2,   w2_bf,   n);

    k_embed<<<M, 256, 0, stream>>>(tok, emb, xf, xbf);
    k_bias<<<dim3(N_, B_), 256, 0, stream>>>(dist, tok, dp_w1, dp_b1, dp_w2, dp_b2, biasf);

    for (int l = 0; l < L_; ++l) {
        // QKV projection -> bf16 [B,N,3E]
        k_gemm<<<dim3(E3_ / 128, M / 128), 256, 0, stream>>>(
            xbf, wqkv_bf + (size_t)l * E3_ * E_, bqkv + l * E3_,
            nullptr, qkvbf, M, E3_, E_, 1);
        // attention
        k_attn<<<dim3(H_, B_), 256, 0, stream>>>(qkvbf, biasf, attnbf);
        // output projection -> f32
        k_gemm<<<dim3(E_ / 128, M / 128), 256, 0, stream>>>(
            attnbf, wo_bf + (size_t)l * E_ * E_, bo + l * E_,
            yf, nullptr, M, E_, E_, 0);
        k_ln<<<M, 256, 0, stream>>>(xf, yf, ln1g + l * E_, ln1b + l * E_, xbf);
        // FFN up + exact GELU -> bf16
        k_gemm<<<dim3(FF_ / 128, M / 128), 256, 0, stream>>>(
            xbf, w1_bf + (size_t)l * FF_ * E_, b1 + l * FF_,
            nullptr, hbf, M, FF_, E_, 2);
        // FFN down -> f32
        k_gemm<<<dim3(E_ / 128, M / 128), 256, 0, stream>>>(
            hbf, w2_bf + (size_t)l * E_ * FF_, b2 + l * E_,
            yf, nullptr, M, E_, FF_, 0);
        k_ln<<<M, 256, 0, stream>>>(xf, yf, ln2g + l * E_, ln2b + l * E_, xbf);
    }

    k_out<<<(B_ * N_ * E_ + 255) / 256, 256, 0, stream>>>(xf, tok, (float*)d_out);
}
